// SinkhornPermutation_27504970564038
// MI455X (gfx1250) — matrix operand order repaired and checked by reading, not compiled, not run on hardware
//
#include <hip/hip_runtime.h>
#include <hip/hip_bf16.h>

// Problem constants (from reference setup_inputs)
#define NN      2048     // permutation size (K and N of the GEMM)
#define MB      4096     // batch rows of x (M of the GEMM)
#define NSAMP   4        // gumbel samples
#define NITERS  21       // sinkhorn iterations
#define EPSV    1e-10f

typedef _Float16 v16h __attribute__((ext_vector_type(16)));
typedef float    v8f  __attribute__((ext_vector_type(8)));
typedef int      vi4  __attribute__((ext_vector_type(4)));

union FragH { v16h h; vi4 q[2]; };

// ---------------------------------------------------------------------------
// CDNA5 async global->LDS copy (ASYNCcnt path), guarded so the file compiles
// on toolchains without the builtin (falls back to load+ds_store).
// The builtin's parameters are address-space-qualified vi4 pointers
// (AS1 source prints as "__device__" in clang diagnostics, AS3 dest as LDS).
// ---------------------------------------------------------------------------
#if defined(__gfx1250__) && __has_builtin(__builtin_amdgcn_global_load_async_to_lds_b128)
#define HAVE_ASYNC_LDS 1
typedef __attribute__((address_space(1))) vi4 as1_vi4;
typedef __attribute__((address_space(3))) vi4 as3_vi4;
#else
#define HAVE_ASYNC_LDS 0
#endif

__device__ __forceinline__ void cp16_g2l(const _Float16* g, _Float16* l) {
#if HAVE_ASYNC_LDS
    // 16B memory -> LDS, no VGPR round trip; tracked by ASYNCcnt.
    __builtin_amdgcn_global_load_async_to_lds_b128(
        (as1_vi4*)(unsigned long long)g,
        (as3_vi4*)(unsigned)(unsigned long long)l,
        0, 0);
#else
    *(vi4*)l = *(const vi4*)g;
#endif
}

__device__ __forceinline__ void wait_async_copies() {
#if HAVE_ASYNC_LDS
#if __has_builtin(__builtin_amdgcn_s_wait_asynccnt)
    __builtin_amdgcn_s_wait_asynccnt(0);
#else
    asm volatile("s_wait_asynccnt 0x0" ::: "memory");
#endif
#endif
}

// ---------------------------------------------------------------------------
// 1) la = log_alpha + gumbel(U)   (temp == 1.0 so the /temp is a no-op)
// ---------------------------------------------------------------------------
__global__ void gumbel_add_kernel(const float* __restrict__ logAlpha,
                                  const float* __restrict__ U,
                                  float* __restrict__ la, long total) {
    const long nn = (long)NN * NN;
    long i = (long)blockIdx.x * blockDim.x + threadIdx.x;
    long stride = (long)gridDim.x * blockDim.x;
    for (; i < total; i += stride) {
        float u = U[i];
        float noise = -__logf(EPSV - __logf(u + EPSV));
        la[i] = logAlpha[i % nn] + noise;
    }
}

// ---------------------------------------------------------------------------
// 2a) column logsumexp (reduce over rows, axis=-2), single-pass online LSE.
//     One thread per (sample, column); reads are fully coalesced across j.
// ---------------------------------------------------------------------------
__global__ void col_lse_kernel(const float* __restrict__ la,
                               float* __restrict__ cl) {
    int j = blockIdx.x * blockDim.x + threadIdx.x;   // 0 .. NSAMP*NN-1
    int s = j >> 11;                                 // / NN
    int c = j & (NN - 1);
    const float* p = la + (size_t)s * NN * NN + c;
    float m = -3.0e38f, sum = 0.0f;
    for (int i = 0; i < NN; ++i) {
        float v = p[(size_t)i * NN];
        float nm = fmaxf(m, v);
        sum = sum * __expf(m - nm) + __expf(v - nm);
        m = nm;
    }
    cl[j] = m + __logf(sum);
}

// ---------------------------------------------------------------------------
// 2b) fused: la -= colLSE (bcast over rows); rowLSE; la -= rowLSE.
//     One 256-thread block per (sample,row); 8 elements/thread kept in regs.
// ---------------------------------------------------------------------------
__global__ __launch_bounds__(256) void row_norm_kernel(float* __restrict__ la,
                                                       const float* __restrict__ cl) {
    int s = blockIdx.x >> 11;
    int r = blockIdx.x & (NN - 1);
    float* rowp = la + ((size_t)s * NN + r) * NN;
    const float* clp = cl + s * NN;
    int t = threadIdx.x, lane = t & 31, wid = t >> 5;

    float v[8];
#pragma unroll
    for (int k = 0; k < 8; ++k) {
        int j = t + k * 256;
        v[k] = rowp[j] - clp[j];
    }
    float m = v[0];
#pragma unroll
    for (int k = 1; k < 8; ++k) m = fmaxf(m, v[k]);
#pragma unroll
    for (int off = 16; off > 0; off >>= 1) m = fmaxf(m, __shfl_xor(m, off, 32));

    __shared__ float redm[8];
    __shared__ float reds[8];
    if (lane == 0) redm[wid] = m;
    __syncthreads();
    float gm = redm[0];
#pragma unroll
    for (int w = 1; w < 8; ++w) gm = fmaxf(gm, redm[w]);

    float ssum = 0.0f;
#pragma unroll
    for (int k = 0; k < 8; ++k) ssum += __expf(v[k] - gm);
#pragma unroll
    for (int off = 16; off > 0; off >>= 1) ssum += __shfl_xor(ssum, off, 32);
    if (lane == 0) reds[wid] = ssum;
    __syncthreads();
    float gs = 0.0f;
#pragma unroll
    for (int w = 0; w < 8; ++w) gs += reds[w];

    float lse = gm + __logf(gs);
#pragma unroll
    for (int k = 0; k < 8; ++k) rowp[t + k * 256] = v[k] - lse;
}

// ---------------------------------------------------------------------------
// 3) pT[s][m][n] = (f16) exp(la[s][n][m])   -- tiled LDS transpose so the
//    GEMM's B operand is K-contiguous (matches WMMA B fragment layout).
// ---------------------------------------------------------------------------
__global__ void exp_transpose_kernel(const float* __restrict__ la,
                                     _Float16* __restrict__ pT) {
    __shared__ float tile[32][33];
    int s = blockIdx.z;
    int n0 = blockIdx.y << 5;   // row of la (K dim)
    int m0 = blockIdx.x << 5;   // col of la (output-N dim)
    const float* src = la + (size_t)s * NN * NN;
#pragma unroll
    for (int i = threadIdx.y; i < 32; i += 8)
        tile[i][threadIdx.x] = src[(size_t)(n0 + i) * NN + m0 + threadIdx.x];
    __syncthreads();
    _Float16* dst = pT + (size_t)s * NN * NN;
#pragma unroll
    for (int i = threadIdx.y; i < 32; i += 8)
        dst[(size_t)(m0 + i) * NN + n0 + threadIdx.x] =
            (_Float16)__expf(tile[threadIdx.x][i]);
}

__global__ void cvt_x_kernel(const float* __restrict__ x,
                             _Float16* __restrict__ x16, long total) {
    long i = (long)blockIdx.x * blockDim.x + threadIdx.x;
    long stride = (long)gridDim.x * blockDim.x;
    for (; i < total; i += stride) x16[i] = (_Float16)x[i];
}

// ---------------------------------------------------------------------------
// 4) out[s] = x16 (4096x2048) @ P[s] (2048x2048), P supplied K-contiguous
//    as pT[s][n_out][k].  128x128 block tile, 8 waves of 32(M)x64(N),
//    v_wmma_f32_16x16x32_f16.  K staged through DOUBLE-BUFFERED padded LDS
//    (stride 40 halves: 16B-aligned b128 reads, 20-bank stride -> conflict
//    free) via async global->LDS copies: one barrier per K-step, tile t+1
//    copies overlap tile t's 8 WMMAs.
// ---------------------------------------------------------------------------
__global__ __launch_bounds__(256) void wmma_gemm_kernel(
        const _Float16* __restrict__ x16,
        const _Float16* __restrict__ pT,
        float* __restrict__ out) {
    const int K = NN;
    __shared__ __align__(16) _Float16 ldsA[2][128 * 40];
    __shared__ __align__(16) _Float16 ldsB[2][128 * 40];

    int tid = threadIdx.x;
    int lane = tid & 31, w = tid >> 5;
    int waveM = w & 3;          // 0..3  -> 32-row slabs
    int waveN = w >> 2;         // 0..1  -> 64-col slabs
    int s  = blockIdx.z;
    int m0 = blockIdx.y * 128;
    int n0 = blockIdx.x * 128;

    const _Float16* gA = x16 + (size_t)m0 * K;
    const _Float16* gB = pT + ((size_t)s * NN + n0) * K;

    v8f acc[2][4];
    v8f zero = {};
#pragma unroll
    for (int i = 0; i < 2; ++i)
#pragma unroll
        for (int j = 0; j < 4; ++j) acc[i][j] = zero;

    const int mrow  = lane & 15;
    const int kb    = (lane >> 4) * 8;    // A fragment K-group (ISA 7.12.2)
    const int kbase = (lane >> 4) * 16;   // unused: B fragments use kb, as A does

    // Staging: 512 chunks of 16B per operand per tile; this thread moves
    // chunks {tid, tid+256}: row = id/4, 8-half column group q = id%4.
    const int r0 = tid >> 2,        q0 = tid & 3;
    const int r1 = (tid + 256) >> 2, q1 = (tid + 256) & 3;

    auto stage = [&](int k0, int buf) {
        cp16_g2l(gA + (size_t)r0 * K + k0 + q0 * 8, &ldsA[buf][r0 * 40 + q0 * 8]);
        cp16_g2l(gA + (size_t)r1 * K + k0 + q1 * 8, &ldsA[buf][r1 * 40 + q1 * 8]);
        cp16_g2l(gB + (size_t)r0 * K + k0 + q0 * 8, &ldsB[buf][r0 * 40 + q0 * 8]);
        cp16_g2l(gB + (size_t)r1 * K + k0 + q1 * 8, &ldsB[buf][r1 * 40 + q1 * 8]);
    };

    stage(0, 0);                          // prologue: tile 0 in flight

    for (int k0 = 0; k0 < K; k0 += 32) {
        int buf = (k0 >> 5) & 1;

        wait_async_copies();              // this wave's tile-k0 copies landed
        __syncthreads();                  // everyone's landed; buf^1 no longer read

        if (k0 + 32 < K) stage(k0 + 32, buf ^ 1);   // overlap next tile
        if (k0 + 64 < K) {                // hint tile after next -> global_prefetch_b8
            __builtin_prefetch(gA + (size_t)r0 * K + k0 + 64 + q0 * 8, 0, 1);
            __builtin_prefetch(gB + (size_t)r0 * K + k0 + 64 + q0 * 8, 0, 1);
        }

        const _Float16* bufA = ldsA[buf];
        const _Float16* bufB = ldsB[buf];

        // ---- build fragments from LDS ----
        FragH a[2], b[4];
#pragma unroll
        for (int tm = 0; tm < 2; ++tm) {
            int m = waveM * 32 + tm * 16 + mrow;
            a[tm].q[0] = *(const vi4*)(bufA + m * 40 + kb);        // K=kb..kb+7
            a[tm].q[1] = *(const vi4*)(bufA + m * 40 + 16 + kb);   // K=16+kb..
        }
#pragma unroll
        for (int tn = 0; tn < 4; ++tn) {
            int n = waveN * 64 + tn * 16 + mrow;
            b[tn].q[0] = *(const vi4*)(bufB + n * 40 + kb);        // K=kb..kb+7
            b[tn].q[1] = *(const vi4*)(bufB + n * 40 + 16 + kb);   // K=16+kb..
        }

        // ---- 8 WMMAs per wave per K-step ----
#pragma unroll
        for (int tm = 0; tm < 2; ++tm)
#pragma unroll
            for (int tn = 0; tn < 4; ++tn)
                acc[tm][tn] = __builtin_amdgcn_wmma_f32_16x16x32_f16(
                    false, a[tm].h, false, b[tn].h,
                    (short)0, acc[tm][tn], false, false);
    }

    // ---- store: C/D layout -> VGPR r holds M = r + 8*(lane>=16), N = lane%16
    float* o = out + ((size_t)s * MB + m0) * NN + n0;
    int nlo = lane & 15;
    int mhi = (lane >> 4) * 8;
#pragma unroll
    for (int tm = 0; tm < 2; ++tm)
#pragma unroll
        for (int tn = 0; tn < 4; ++tn) {
            int gn = waveN * 64 + tn * 16 + nlo;
#pragma unroll
            for (int r = 0; r < 8; ++r) {
                int gm = waveM * 32 + tm * 16 + mhi + r;
                o[(size_t)gm * NN + gn] = acc[tm][tn][r];
            }
        }
}

// ---------------------------------------------------------------------------
// Host side: ws layout (total ~112 MB):
//   la   : NSAMP*NN*NN f32  = 64 MiB   (fits in MI455X 192 MB L2 -> sinkhorn
//                                        iterations run at L2, not HBM, BW)
//   cl   : NSAMP*NN    f32  = 32 KiB
//   pT   : NSAMP*NN*NN f16  = 32 MiB
//   x16  : MB*NN       f16  = 16 MiB
// ---------------------------------------------------------------------------
extern "C" void kernel_launch(void* const* d_in, const int* in_sizes, int n_in,
                              void* d_out, int out_size, void* d_ws, size_t ws_size,
                              hipStream_t stream) {
    const float* x        = (const float*)d_in[0];   // (4096, 2048)
    const float* logAlpha = (const float*)d_in[1];   // (2048, 2048)
    const float* U        = (const float*)d_in[2];   // (4, 2048, 2048)
    float* out = (float*)d_out;                      // (4, 4096, 2048)

    const size_t nn = (size_t)NN * NN;
    char* ws = (char*)d_ws;
    float*    la  = (float*)ws;
    float*    cl  = (float*)(ws + NSAMP * nn * sizeof(float));
    _Float16* pT  = (_Float16*)(ws + NSAMP * nn * sizeof(float)
                                   + NSAMP * NN * sizeof(float));
    _Float16* x16 = (_Float16*)((char*)pT + NSAMP * nn * sizeof(_Float16));

    long totalLA = (long)NSAMP * nn;
    gumbel_add_kernel<<<4096, 256, 0, stream>>>(logAlpha, U, la, totalLA);

    for (int it = 0; it < NITERS; ++it) {
        col_lse_kernel<<<(NSAMP * NN) / 256, 256, 0, stream>>>(la, cl);
        row_norm_kernel<<<NSAMP * NN, 256, 0, stream>>>(la, cl);
    }

    exp_transpose_kernel<<<dim3(NN / 32, NN / 32, NSAMP), dim3(32, 8), 0, stream>>>(la, pT);
    cvt_x_kernel<<<2048, 256, 0, stream>>>(x, x16, (long)MB * NN);

    wmma_gemm_kernel<<<dim3(NN / 128, MB / 128, NSAMP), 256, 0, stream>>>(x16, pT, out);
}
